// GCNAttention_64132451664025
// MI455X (gfx1250) — matrix operand order repaired and checked by reading, not compiled, not run on hardware
//
#include <hip/hip_runtime.h>
#include <math.h>

// ---------------------------------------------------------------------------
// GCN + single-head graph attention forward, MI455X (gfx1250, wave32, WMMA).
// Heavy GEMMs run as bf16 WMMA (v_wmma_f32_16x16x32_bf16) with fp32 accum.
// adj entries are exactly 0/1 -> exact in bf16.
// fp32 -> bf16 conversion via native __bf16 casts (v_cvt_pk_bf16_f32).
// FULLN template flag removes all bounds-check divergence from the
// full-width (Nb%64==0) instantiations' inner loops.
// ---------------------------------------------------------------------------

#define N_NODES 8192
#define NFEAT   512
#define NHID    256
#define NCLASS  16
#define ALPHA   0.01f

typedef __bf16 v16bf __attribute__((ext_vector_type(16)));
typedef __bf16 v2bf  __attribute__((ext_vector_type(2)));
typedef float  v8f   __attribute__((ext_vector_type(8)));

__device__ __forceinline__ unsigned int pack2(float a, float b) {
  v2bf p;
  p[0] = (__bf16)a;            // backend: v_cvt_pk_bf16_f32
  p[1] = (__bf16)b;
  return __builtin_bit_cast(unsigned int, p);
}

union FragU { uint4 u[2]; v16bf v; };

// ---------------------------------------------------------------------------
// Tiled WMMA GEMM:  C[M,Nb] = epi( A[M,K] * B[K,Nb] (+bias) )
// AKIND: 0 = fp32 source, 1 = int32 0/1 source (adjacency)
// EPI:   0 = none, 1 = +bias then ReLU, 2 = ELU, 3 = +bias
// FULLN: Nb is a multiple of 64 -> no bounds checks anywhere
// Block: 256 threads (8 waves). Tile 128x64, K-step 32.
// Wave w computes rows [16w,16w+16) x all 64 cols (4 accumulators).
// ---------------------------------------------------------------------------
template<int AKIND, int EPI, bool FULLN>
__global__ __launch_bounds__(256, 2)
void wmma_gemm_bf16(const void* __restrict__ Ap, const float* __restrict__ Bp,
                    const float* __restrict__ bias, float* __restrict__ C,
                    int M, int Nb, int K)
{
  // LDS: A tile 128x32 bf16 (stride 40 -> 80B rows, 16B aligned, no bank
  // conflicts across 16-lane fragment loads); B tile stored transposed
  // BsT[n][k] so fragment loads are contiguous b128s.
  __shared__ __align__(16) unsigned short As[128 * 40];
  __shared__ __align__(16) unsigned short Bs[64 * 40];

  const int tid  = threadIdx.x;
  const int lane = tid & 31;
  const int wv   = tid >> 5;

  const int rowBase = blockIdx.y * 128;
  const int cb      = blockIdx.x * 64;

  // A stage coords: 2 threads per row, 16 contiguous elems each
  const int ar  = tid >> 1;
  const int acg = (tid & 1) * 16;
  // B stage coords: thread handles K-pair kp (k=2kp,2kp+1) x 4 cols
  const int kp  = tid & 15;
  const int bn0 = (tid >> 4) * 4;

  v8f acc[4];
  #pragma unroll
  for (int f = 0; f < 4; ++f)
    #pragma unroll
    for (int q = 0; q < 8; ++q) acc[f][q] = 0.0f;

  // Fragment coords (ISA 7.12.2 layouts)
  const int arow = wv * 16 + (lane & 15);
  const int akb  = (lane < 16) ? 0 : 8;    // A: K groups {0..7,16..23}/{8..15,24..31}
  const int bkb  = (lane < 16) ? 0 : 8;    // B: K groups {0..7,16..23}/{8..15,24..31}

  for (int k0 = 0; k0 < K; k0 += 32) {
    __syncthreads();
    // ---- stage A tile (128 x 32) fp32/int32 -> bf16 ----
    {
      unsigned int u[8];
      if (AKIND == 0) {
        const float* a = (const float*)Ap + (size_t)(rowBase + ar) * K + k0 + acg;
        #pragma unroll
        for (int q = 0; q < 4; ++q) {
          float4 f = ((const float4*)a)[q];
          u[2*q]   = pack2(f.x, f.y);
          u[2*q+1] = pack2(f.z, f.w);
        }
      } else {
        const int* a = (const int*)Ap + (size_t)(rowBase + ar) * K + k0 + acg;
        #pragma unroll
        for (int q = 0; q < 4; ++q) {
          int4 f = ((const int4*)a)[q];
          u[2*q]   = (f.x ? 0x3F80u : 0u) | ((f.y ? 0x3F80u : 0u) << 16);
          u[2*q+1] = (f.z ? 0x3F80u : 0u) | ((f.w ? 0x3F80u : 0u) << 16);
        }
      }
      uint4* dst = (uint4*)&As[ar * 40 + acg];
      dst[0] = make_uint4(u[0], u[1], u[2], u[3]);
      dst[1] = make_uint4(u[4], u[5], u[6], u[7]);
    }
    // ---- stage B tile (32 x 64) fp32 -> bf16, transposed BsT[n][k] ----
    {
      const int gk = k0 + 2 * kp;
      float c0[4], c1[4];
      if (FULLN || (cb + bn0 + 3 < Nb)) {
        float4 r0 = *(const float4*)(Bp + (size_t)gk       * Nb + cb + bn0);
        float4 r1 = *(const float4*)(Bp + (size_t)(gk + 1) * Nb + cb + bn0);
        c0[0] = r0.x; c0[1] = r0.y; c0[2] = r0.z; c0[3] = r0.w;
        c1[0] = r1.x; c1[1] = r1.y; c1[2] = r1.z; c1[3] = r1.w;
      } else {
        #pragma unroll
        for (int j = 0; j < 4; ++j) {
          const int c = cb + bn0 + j;
          c0[j] = (c < Nb) ? Bp[(size_t)gk       * Nb + c] : 0.0f;
          c1[j] = (c < Nb) ? Bp[(size_t)(gk + 1) * Nb + c] : 0.0f;
        }
      }
      #pragma unroll
      for (int j = 0; j < 4; ++j)   // low bf16 = even K, high = odd K
        *(unsigned int*)&Bs[(bn0 + j) * 40 + 2 * kp] = pack2(c0[j], c1[j]);
    }

    // prefetch next K-step's global lines (global_prefetch_b8) to hide HBM
    // latency of the 268MB adjacency/attention streams behind WMMA work
    if (k0 + 32 < K) {
      if (AKIND == 0)
        __builtin_prefetch((const float*)Ap + (size_t)(rowBase + ar) * K + k0 + 32 + acg, 0, 0);
      else
        __builtin_prefetch((const int*)Ap + (size_t)(rowBase + ar) * K + k0 + 32 + acg, 0, 0);
      __builtin_prefetch(Bp + (size_t)(k0 + 32 + 2 * kp) * Nb + cb + bn0, 0, 0);
    }

    __syncthreads();
    // ---- fragments + WMMA (EXEC all-1s here: no divergence) ----
    FragU fa;
    fa.u[0] = *(const uint4*)&As[arow * 40 + akb];
    fa.u[1] = *(const uint4*)&As[arow * 40 + akb + 16];
    #pragma unroll
    for (int nf = 0; nf < 4; ++nf) {
      const int bc = nf * 16 + (lane & 15);
      FragU fb;
      fb.u[0] = *(const uint4*)&Bs[bc * 40 + bkb];
      fb.u[1] = *(const uint4*)&Bs[bc * 40 + bkb + 16];
      acc[nf] = __builtin_amdgcn_wmma_f32_16x16x32_bf16(
          false, fa.v, false, fb.v, (short)0, acc[nf], false, false);
    }
  }

  // ---- epilogue + store (C/D layout: VGPR q <-> row q / q+8) ----
  const int mb    = rowBase + wv * 16 + ((lane >= 16) ? 8 : 0);
  const int nlane = lane & 15;
  #pragma unroll
  for (int nf = 0; nf < 4; ++nf) {
    const int col = cb + nf * 16 + nlane;
    if (FULLN || (col < Nb)) {
      const float bvi = (EPI == 1 || EPI == 3) ? bias[col] : 0.0f;
      #pragma unroll
      for (int q = 0; q < 8; ++q) {
        float v = acc[nf][q];
        if      (EPI == 1) { v += bvi; v = v > 0.0f ? v : 0.0f; }
        else if (EPI == 2) { v = v > 0.0f ? v : (__expf(v) - 1.0f); }
        else if (EPI == 3) { v += bvi; }
        C[(size_t)(mb + q) * Nb + col] = v;
      }
    }
  }
}

// ---------------------------------------------------------------------------
// f1[i] = Wh[i,:].a[:256], f2[i] = Wh[i,:].a[256:]  — one wave per row
// ---------------------------------------------------------------------------
__global__ __launch_bounds__(256)
void f1f2_kernel(const float* __restrict__ Wh, const float* __restrict__ avec,
                 float* __restrict__ f1, float* __restrict__ f2)
{
  const int gw   = (blockIdx.x * blockDim.x + threadIdx.x) >> 5;  // row
  const int lane = threadIdx.x & 31;
  if (gw >= N_NODES) return;
  const float* row = Wh + (size_t)gw * NHID;
  float s1 = 0.0f, s2 = 0.0f;
  #pragma unroll
  for (int j = lane; j < NHID; j += 32) {
    const float w = row[j];
    s1 += w * avec[j];
    s2 += w * avec[NHID + j];
  }
  #pragma unroll
  for (int off = 16; off; off >>= 1) {
    s1 += __shfl_down(s1, off);
    s2 += __shfl_down(s2, off);
  }
  if (lane == 0) { f1[gw] = s1; f2[gw] = s2; }
}

// ---------------------------------------------------------------------------
// Masked row softmax: att[i,j] = softmax_j( adj>0 ? leaky(f1[i]+f2[j]) : -inf )
// One block per row; the 32KB score row lives in LDS (320KB/WGP on CDNA5).
// ---------------------------------------------------------------------------
__global__ __launch_bounds__(256)
void att_softmax(const int* __restrict__ adj, const float* __restrict__ f1,
                 const float* __restrict__ f2, float* __restrict__ att)
{
  __shared__ float e[N_NODES];
  __shared__ float red[8];
  const int i   = blockIdx.x;
  const int tid = threadIdx.x;
  const int lane = tid & 31, wv = tid >> 5;
  const float f1i = f1[i];
  const int* arow = adj + (size_t)i * N_NODES;

  float lmax = -INFINITY;
  for (int j = tid; j < N_NODES; j += 256) {
    const int a = arow[j];
    float v;
    if (a != 0) { const float t = f1i + f2[j]; v = t > 0.0f ? t : ALPHA * t; }
    else          v = -INFINITY;
    e[j] = v;
    lmax = fmaxf(lmax, v);
  }
  #pragma unroll
  for (int off = 16; off; off >>= 1) lmax = fmaxf(lmax, __shfl_down(lmax, off));
  if (lane == 0) red[wv] = lmax;
  __syncthreads();
  if (tid == 0) {
    float m = red[0];
    #pragma unroll
    for (int k = 1; k < 8; ++k) m = fmaxf(m, red[k]);
    red[0] = m;
  }
  __syncthreads();
  const float bmax = red[0];
  __syncthreads();

  float lsum = 0.0f;
  for (int j = tid; j < N_NODES; j += 256) {
    const float v = e[j];
    const float ex = (v <= -1.0e30f) ? 0.0f : __expf(v - bmax);
    e[j] = ex;
    lsum += ex;
  }
  #pragma unroll
  for (int off = 16; off; off >>= 1) lsum += __shfl_down(lsum, off);
  if (lane == 0) red[wv] = lsum;
  __syncthreads();
  if (tid == 0) {
    float s = 0.0f;
    #pragma unroll
    for (int k = 0; k < 8; ++k) s += red[k];
    red[0] = s;
  }
  __syncthreads();
  const float inv = 1.0f / red[0];

  float* orow = att + (size_t)i * N_NODES;
  for (int j = tid; j < N_NODES; j += 256) orow[j] = e[j] * inv;
}

// ---------------------------------------------------------------------------
// d_out[0:131072)            = sigmoid(out)
// d_out[concat .. +8192*272) = [h1 | out]
// ---------------------------------------------------------------------------
__global__ __launch_bounds__(256)
void finalize_kernel(const float* __restrict__ out_raw, const float* __restrict__ h1,
                     float* __restrict__ d_out, size_t concat_off)
{
  const int i   = blockIdx.x;
  const int tid = threadIdx.x;
  const float* orow = out_raw + (size_t)i * NCLASS;
  if (tid < NCLASS)
    d_out[(size_t)i * NCLASS + tid] = 1.0f / (1.0f + __expf(-orow[tid]));
  float* crow = d_out + concat_off + (size_t)i * (NHID + NCLASS);
  for (int c = tid; c < NHID + NCLASS; c += 256)
    crow[c] = (c < NHID) ? h1[(size_t)i * NHID + c] : orow[c - NHID];
}

// ---------------------------------------------------------------------------
extern "C" void kernel_launch(void* const* d_in, const int* in_sizes, int n_in,
                              void* d_out, int out_size, void* d_ws, size_t ws_size,
                              hipStream_t stream) {
  (void)in_sizes; (void)n_in; (void)out_size; (void)ws_size;

  const float* x     = (const float*)d_in[0];
  const int*   adj   = (const int*)  d_in[1];
  const float* W1    = (const float*)d_in[2];
  const float* b1    = (const float*)d_in[3];
  const float* Wa    = (const float*)d_in[4];
  const float* a_vec = (const float*)d_in[5];
  const float* W3    = (const float*)d_in[6];
  const float* b3    = (const float*)d_in[7];
  float* out = (float*)d_out;

  // workspace layout (fp32), ~34.7 MB total
  float* XW1     = (float*)d_ws;                 // 8192*256
  float* h1      = XW1     + (size_t)N_NODES * NHID;
  float* Wh      = h1      + (size_t)N_NODES * NHID;
  float* a_out   = Wh      + (size_t)N_NODES * NHID;
  float* f1      = a_out   + (size_t)N_NODES * NHID;
  float* f2      = f1      + N_NODES;
  float* aW3     = f2      + N_NODES;            // 8192*16
  float* out_raw = aW3     + (size_t)N_NODES * NCLASS;

  const size_t att_off    = (size_t)N_NODES * NCLASS;               // 131072
  const size_t concat_off = att_off + (size_t)N_NODES * N_NODES;    // 67239936
  float* att = out + att_off;

  const dim3 blk(256);
  const dim3 gFull(NHID / 64, N_NODES / 128);    // (4, 64)
  const dim3 gNarrow(1, N_NODES / 128);          // (1, 64), Nb = 16

  // 1) XW1 = x @ W1                       [8192,512]x[512,256]
  wmma_gemm_bf16<0, 0, true><<<gFull, blk, 0, stream>>>(x, W1, nullptr, XW1,
                                                        N_NODES, NHID, NFEAT);
  // 2) h1 = relu(adj @ XW1 + b1)          [8192,8192]x[8192,256]
  wmma_gemm_bf16<1, 1, true><<<gFull, blk, 0, stream>>>(adj, XW1, b1, h1,
                                                        N_NODES, NHID, N_NODES);
  // 3) Wh = h1 @ Wa                       [8192,256]x[256,256]
  wmma_gemm_bf16<0, 0, true><<<gFull, blk, 0, stream>>>(h1, Wa, nullptr, Wh,
                                                        N_NODES, NHID, NHID);
  // 4) f1, f2 (attention logits halves)
  f1f2_kernel<<<N_NODES / 8, blk, 0, stream>>>(Wh, a_vec, f1, f2);
  // 5) attention rows -> d_out (masked softmax)
  att_softmax<<<N_NODES, blk, 0, stream>>>(adj, f1, f2, att);
  // 6) a_out = elu(attention @ Wh)        [8192,8192]x[8192,256]
  wmma_gemm_bf16<0, 2, true><<<gFull, blk, 0, stream>>>(att, Wh, nullptr, a_out,
                                                        N_NODES, NHID, N_NODES);
  // 7) aW3 = a_out @ W3                   [8192,256]x[256,16]
  wmma_gemm_bf16<0, 0, false><<<gNarrow, blk, 0, stream>>>(a_out, W3, nullptr, aW3,
                                                           N_NODES, NCLASS, NHID);
  // 8) out = adj @ aW3 + b3               [8192,8192]x[8192,16]
  wmma_gemm_bf16<1, 3, false><<<gNarrow, blk, 0, stream>>>(adj, aW3, b3, out_raw,
                                                           N_NODES, NCLASS, N_NODES);
  // 9) sigmoid head + concat
  finalize_kernel<<<N_NODES, blk, 0, stream>>>(out_raw, h1, out, concat_off);
}
